// ImprovedGraphLSTMCell_32933809225898
// MI455X (gfx1250) — matrix operand order repaired and checked by reading, not compiled, not run on hardware
//
#include <hip/hip_runtime.h>
#include <hip/hip_bf16.h>

// Problem constants (from reference): N=8192, K=8, H=X=512
#define NND   8192
#define KNB   8
#define HH    512
#define XD    512
#define AK    1536       // K-dim of fused pre-GEMM: [x(512) | hsum0(512) | hsum1(512)]
#define BC    2048       // cols of fused pre-GEMM: [iou(1536) | f(512)]
#define UK    1024       // K-dim of f-GEMM (2H)
#define APAD  (AK + 8)   // LDS row stride (bf16) -> 3088 B: conflict-free b128 reads
#define UPAD  (UK + 8)   // LDS row stride (bf16) -> 2064 B: conflict-free b128 reads

typedef float  v8f   __attribute__((ext_vector_type(8)));
typedef __bf16 v8bf  __attribute__((ext_vector_type(8)));
typedef __bf16 v16bf __attribute__((ext_vector_type(16)));
typedef int    v4i   __attribute__((ext_vector_type(4)));

static __device__ __forceinline__ float sig_(float x) {
  return 1.0f / (1.0f + __expf(-x));
}
static __device__ __forceinline__ float tanh_(float x) {
  float e = __expf(-2.0f * x);
  return (1.0f - e) / (1.0f + e);
}

// Kernel 1: A-buffer for fused node GEMM: [bf16(x) | bf16(sum (1-e)h) | bf16(sum e h)]
__global__ void __launch_bounds__(256) prep_abuf(
    const float* __restrict__ x, const float* __restrict__ h_mail,
    const float* __restrict__ edge_label, __bf16* __restrict__ Abuf) {
  const int n = blockIdx.x;
  for (int j = threadIdx.x; j < HH; j += blockDim.x) {
    Abuf[(size_t)n * AK + j] = (__bf16)x[(size_t)n * XD + j];
    float s0 = 0.0f, s1 = 0.0f;
#pragma unroll
    for (int k = 0; k < KNB; ++k) {
      float e = edge_label[n * KNB + k];
      float h = h_mail[((size_t)n * KNB + k) * HH + j];
      s1 += e * h;
      s0 += (1.0f - e) * h;
    }
    Abuf[(size_t)n * AK + HH + j]     = (__bf16)s0;
    Abuf[(size_t)n * AK + 2 * HH + j] = (__bf16)s1;
  }
}

// Kernel 2: combined transposed bf16 weights BT[c][k], c in [0,2048), k in [0,1536)
__global__ void __launch_bounds__(256) prep_weights(
    const float* __restrict__ W_iou, const float* __restrict__ W_f,
    const float* __restrict__ U_iou, const float* __restrict__ U_f,
    __bf16* __restrict__ BT) {
  int idx = blockIdx.x * blockDim.x + threadIdx.x;
  if (idx >= BC * AK) return;
  int c = idx / AK, k = idx % AK;
  float v;
  if (k < XD) {
    v = (c < 3 * HH) ? W_iou[(size_t)k * (3 * HH) + c]
                     : W_f[(size_t)k * HH + (c - 3 * HH)];
  } else {
    int kk = k - XD;
    v = (c < 3 * HH) ? U_iou[(size_t)kk * (3 * HH) + c]
                     : U_f[(size_t)kk * HH + (c - 3 * HH)];
  }
  BT[(size_t)c * AK + k] = (__bf16)v;
}

static __device__ __forceinline__ v16bf make_afrag(const __bf16* p) {
  // A 16x32 bf16 fragment: 16 contiguous k (two 8-elem chunks at k and k+8)
  v8bf lo = *(const v8bf*)p;
  v8bf hi = *(const v8bf*)(p + 8);
  v16bf f;
#pragma unroll
  for (int i = 0; i < 8; ++i) { f[i] = lo[i]; f[i + 8] = hi[i]; }
  return f;
}

// Kernel 3: pre = [x|hsum0|hsum1] (8192x1536) @ BT^T (1536x2048), f32 out.
// Block: 64 rows x 512 cols. 8 waves; each wave: 4 row-tiles x 4 col-tiles.
// Each B fragment is reused by 4 WMMAs (16 WMMA per 4 B-loads per k-step).
__global__ void __launch_bounds__(256) gemm_pre(
    const __bf16* __restrict__ Abuf, const __bf16* __restrict__ BT,
    float* __restrict__ pre) {
  __shared__ __align__(16) __bf16 As[64 * APAD];   // 197632 B (< 320KB WGP LDS)
  const int tid = threadIdx.x;
  const int rb = blockIdx.y, cb = blockIdx.x;
  const size_t abase = (size_t)rb * 64 * AK;
  // A tile: 64 rows x 1536 bf16, copied in 16B chunks into padded rows
  for (int idx = tid; idx < 64 * (AK / 8); idx += 256) {
    int row = idx / (AK / 8), c8 = idx % (AK / 8);
    *(v4i*)&As[row * APAD + c8 * 8] = *(const v4i*)&Abuf[abase + row * AK + c8 * 8];
  }
  __syncthreads();

  const int wave = tid >> 5, lane = tid & 31;
  const int lrow = lane & 15;
  const int ka  = (lane >= 16) ? 8 : 0;    // lane-half C row offset (A uses kbv)
  const int kbv = (lane >= 16) ? 16 : 0;   // B lane-half k offset
  const int c0w = cb * 512 + wave * 64;

  v8f acc[4][4] = {};                       // [row-tile][col-tile]
  for (int kb = 0; kb < AK; kb += 32) {
    v16bf af[4];
#pragma unroll
    for (int r = 0; r < 4; ++r)
      af[r] = make_afrag(&As[(r * 16 + lrow) * APAD + kb + kbv]);
#pragma unroll
    for (int t = 0; t < 4; ++t) {
      const __bf16* bp = &BT[(size_t)(c0w + t * 16 + lrow) * AK + kb + kbv];
      v16bf bfr = *(const v16bf*)bp;        // 16 contiguous k per lane
#pragma unroll
      for (int r = 0; r < 4; ++r)
        acc[r][t] = __builtin_amdgcn_wmma_f32_16x16x32_bf16(
            false, af[r], false, bfr, (short)0, acc[r][t], false, false);
    }
  }
#pragma unroll
  for (int r = 0; r < 4; ++r)
#pragma unroll
    for (int t = 0; t < 4; ++t) {
      int col = c0w + t * 16 + lrow;
#pragma unroll
      for (int v = 0; v < 8; ++v) {
        int row = rb * 64 + r * 16 + v + ka; // C layout: M = v + (lane>=16 ? 8 : 0)
        pre[(size_t)row * BC + col] = acc[r][t][v];
      }
    }
}

// Kernel 4: f-GEMM + full fused epilogue.
// Block = 64 (n,k) rows = nodes rb*8..rb*8+7; A = routed h_mail in LDS (64x1024 bf16);
// B = U_f^T slice of BT; epilogue lane-local: acc VGPR v == neighbor k of one node.
__global__ void __launch_bounds__(256) gemm_f_final(
    const __bf16* __restrict__ BT, const float* __restrict__ pre,
    const float* __restrict__ h_mail, const float* __restrict__ c_mail,
    const float* __restrict__ edge_label,
    const float* __restrict__ b_iou, const float* __restrict__ b_f,
    float* __restrict__ out) {
  __shared__ __align__(16) __bf16 Hs[64 * UPAD];   // 132096 B
  const int tid = threadIdx.x;
  const int rb = blockIdx.x;
  for (int idx = tid; idx < 64 * HH; idx += 256) {
    int row = idx >> 9, j = idx & (HH - 1);
    int gr = rb * 64 + row;                 // global (n,k) row
    float e = edge_label[gr];
    float h = h_mail[(size_t)gr * HH + j];
    int off = (e > 0.5f) ? HH : 0;          // route into half selected by label
    Hs[row * UPAD + off + j]        = (__bf16)h;
    Hs[row * UPAD + (off ^ HH) + j] = (__bf16)0.0f;
  }
  __syncthreads();

  const int wave = tid >> 5, lane = tid & 31;
  const int lrow = lane & 15;
  const int ka  = (lane >= 16) ? 8 : 0;
  const int kbv = (lane >= 16) ? 16 : 0;
  const int c0w = wave * 64;
  const __bf16* BU = BT + (size_t)(3 * HH) * AK + XD;  // U_f^T: cols 1536.., k 512..

  v8f acc[4][4] = {};                       // [row-tile][col-tile]
  for (int kb = 0; kb < UK; kb += 32) {
    v16bf af[4];
#pragma unroll
    for (int r = 0; r < 4; ++r)
      af[r] = make_afrag(&Hs[(r * 16 + lrow) * UPAD + kb + kbv]);
#pragma unroll
    for (int t = 0; t < 4; ++t) {
      const __bf16* bp = BU + (size_t)(c0w + t * 16 + lrow) * AK + kb + kbv;
      v16bf bfr = *(const v16bf*)bp;
#pragma unroll
      for (int r = 0; r < 4; ++r)
        acc[r][t] = __builtin_amdgcn_wmma_f32_16x16x32_bf16(
            false, af[r], false, bfr, (short)0, acc[r][t], false, false);
    }
  }

  // Epilogue: per row-tile, lanes 0-15 own its even node, lanes 16-31 the odd node;
  // acc VGPR v == neighbor k, so c_agg reduces entirely in-register.
  const int nloc = (lane >= 16) ? 1 : 0;
#pragma unroll
  for (int r = 0; r < 4; ++r) {
    const int node = rb * 8 + r * 2 + nloc;
#pragma unroll
    for (int t = 0; t < 4; ++t) {
      int col = c0w + t * 16 + lrow;
      float wfx = pre[(size_t)node * BC + 3 * HH + col];
      float bfv = b_f[col];
      float cagg = 0.0f;
#pragma unroll
      for (int v = 0; v < 8; ++v) {
        int gr = rb * 64 + r * 16 + nloc * 8 + v;
        float f = sig_(acc[r][t][v] + wfx + bfv);
        cagg += f * c_mail[(size_t)gr * HH + col];
      }
      float iv = pre[(size_t)node * BC + col]          + b_iou[col];
      float ov = pre[(size_t)node * BC + HH + col]     + b_iou[HH + col];
      float uv = pre[(size_t)node * BC + 2 * HH + col] + b_iou[2 * HH + col];
      float c  = sig_(iv) * tanh_(uv) + cagg;
      float h  = sig_(ov) * tanh_(c);
      out[(size_t)node * HH + col] = h;
      out[(size_t)NND * HH + (size_t)node * HH + col] = c;
    }
  }
}

extern "C" void kernel_launch(void* const* d_in, const int* in_sizes, int n_in,
                              void* d_out, int out_size, void* d_ws, size_t ws_size,
                              hipStream_t stream) {
  (void)in_sizes; (void)n_in; (void)out_size; (void)ws_size;
  const float* x          = (const float*)d_in[0];
  const float* h_mail     = (const float*)d_in[1];
  const float* c_mail     = (const float*)d_in[2];
  const float* edge_label = (const float*)d_in[3];
  const float* W_iou      = (const float*)d_in[4];
  const float* W_f        = (const float*)d_in[5];
  const float* U_iou      = (const float*)d_in[6];
  const float* U_f        = (const float*)d_in[7];
  const float* b_iou      = (const float*)d_in[8];
  const float* b_f        = (const float*)d_in[9];
  float* out = (float*)d_out;

  char* ws = (char*)d_ws;
  __bf16* Abuf = (__bf16*)ws;                                  // 8192*1536*2  = 25165824 B
  __bf16* BT   = (__bf16*)(ws + 25165824);                     // 2048*1536*2  =  6291456 B
  float*  pre  = (float*)(ws + 25165824 + 6291456);            // 8192*2048*4  = 67108864 B

  prep_abuf<<<NND, 256, 0, stream>>>(x, h_mail, edge_label, Abuf);
  prep_weights<<<(BC * AK) / 256, 256, 0, stream>>>(W_iou, W_f, U_iou, U_f, BT);
  gemm_pre<<<dim3(4, NND / 64), 256, 0, stream>>>(Abuf, BT, pre);
  gemm_f_final<<<(NND * KNB) / 64, 256, 0, stream>>>(BT, pre, h_mail, c_mail,
                                                     edge_label, b_iou, b_f, out);
}
